// NonLocalBlock_68247030334369
// MI455X (gfx1250) — compile-verified, matrix operand order checked by reading, not run on hardware
//
#include <hip/hip_runtime.h>
#include <hip/hip_bf16.h>
#include <stdint.h>

// ---- problem constants (match reference) ----
#define BATCH 8
#define SEQ   1024
#define IDIM  1024
#define ODIM  1024
#define NH    16
#define HD    64
#define KHALF 7         // window = +/-7 -> 15 taps
#define NTAP  15

typedef __attribute__((ext_vector_type(16))) __bf16 v16bf;
typedef __attribute__((ext_vector_type(8)))  __bf16 v8bf;
typedef __attribute__((ext_vector_type(8)))  float  v8f;
typedef __attribute__((ext_vector_type(4)))  float  f4vec;

static __device__ inline uint16_t f32_to_bf16_bits(float f) {
    uint32_t u = __builtin_bit_cast(uint32_t, f);
    u += 0x7FFFu + ((u >> 16) & 1u);   // round-to-nearest-even
    return (uint16_t)(u >> 16);
}

static __device__ inline uint32_t pk2bf(float a, float b) {
    return (uint32_t)f32_to_bf16_bits(a) | ((uint32_t)f32_to_bf16_bits(b) << 16);
}

static __device__ inline float wave_reduce_sum(float v) {
    v += __shfl_xor(v, 16, 32);
    v += __shfl_xor(v, 8, 32);
    v += __shfl_xor(v, 4, 32);
    v += __shfl_xor(v, 2, 32);
    v += __shfl_xor(v, 1, 32);
    return v;
}

// =====================================================================
// Tiled bf16 WMMA GEMM:  C[M=8192, N] = A[8192,1024] * W[1024,N] + bias
// Block tile 128x64, K-step 32, 256 threads = 8 waves (4x2 wave grid),
// each wave computes a 32x32 tile = 2x2 v_wmma_f32_16x16x32_bf16 frags.
// LDS double-buffered: 1 barrier / K-step, global loads overlap WMMA.
// =====================================================================
#define BM 128
#define BN 64
#define BK 32
#define LDS_STRIDE 40   // halfs per row (32 data + 8 pad), 80B: 16B aligned
#define NKSTEPS (IDIM / BK)

template<int N>
__global__ __launch_bounds__(256) void gemm_bf16_wmma(
    const float* __restrict__ A, const float* __restrict__ W,
    const float* __restrict__ bias, float* __restrict__ C)
{
    __shared__ __align__(16) uint16_t As[2][BM * LDS_STRIDE];   // [row][k]
    __shared__ __align__(16) uint16_t Bs[2][BN * LDS_STRIDE];   // [col][k]

    const int tid  = threadIdx.x;
    const int lane = tid & 31;
    const int wave = tid >> 5;          // 0..7
    const int waveM = wave >> 1;        // 0..3
    const int waveN = wave & 1;         // 0..1

    const int m0 = blockIdx.y * BM;
    const int n0 = blockIdx.x * BN;

    v8f acc[2][2];
    #pragma unroll
    for (int f = 0; f < 2; ++f)
        #pragma unroll
        for (int g = 0; g < 2; ++g)
            #pragma unroll
            for (int r = 0; r < 8; ++r) acc[f][g][r] = 0.0f;

    // staging roles
    const int aRow = tid >> 1;            // 0..127
    const int aKg  = (tid & 1) * 16;      // 0 or 16
    const int bK   = tid >> 3;            // 0..31
    const int bNg  = (tid & 7) * 8;       // 0,8,..,56

    const float* aSrcBase = &A[(size_t)(m0 + aRow) * IDIM + aKg];
    const float* wSrcBase = &W[(size_t)bK * N + n0 + bNg];

    const int rA  = lane & 15;
    const int kh8 = (lane >> 4) * 8;      // 0 or 8

    // ---- prologue: stage tile 0 into buffer 0 ----
    {
        const float4* sa = (const float4*)aSrcBase;
        float4 p0 = sa[0], p1 = sa[1], p2 = sa[2], p3 = sa[3];
        uint4 u0 = { pk2bf(p0.x, p0.y), pk2bf(p0.z, p0.w),
                     pk2bf(p1.x, p1.y), pk2bf(p1.z, p1.w) };
        uint4 u1 = { pk2bf(p2.x, p2.y), pk2bf(p2.z, p2.w),
                     pk2bf(p3.x, p3.y), pk2bf(p3.z, p3.w) };
        *(uint4*)&As[0][aRow * LDS_STRIDE + aKg]     = u0;
        *(uint4*)&As[0][aRow * LDS_STRIDE + aKg + 8] = u1;

        const float4* sw = (const float4*)wSrcBase;
        float4 q0 = sw[0], q1 = sw[1];
        Bs[0][(bNg + 0) * LDS_STRIDE + bK] = f32_to_bf16_bits(q0.x);
        Bs[0][(bNg + 1) * LDS_STRIDE + bK] = f32_to_bf16_bits(q0.y);
        Bs[0][(bNg + 2) * LDS_STRIDE + bK] = f32_to_bf16_bits(q0.z);
        Bs[0][(bNg + 3) * LDS_STRIDE + bK] = f32_to_bf16_bits(q0.w);
        Bs[0][(bNg + 4) * LDS_STRIDE + bK] = f32_to_bf16_bits(q1.x);
        Bs[0][(bNg + 5) * LDS_STRIDE + bK] = f32_to_bf16_bits(q1.y);
        Bs[0][(bNg + 6) * LDS_STRIDE + bK] = f32_to_bf16_bits(q1.z);
        Bs[0][(bNg + 7) * LDS_STRIDE + bK] = f32_to_bf16_bits(q1.w);
    }
    __syncthreads();

    for (int kb = 0; kb < NKSTEPS; ++kb) {
        const int cur = kb & 1;
        const bool hasNext = (kb + 1) < NKSTEPS;

        // ---- issue next tile's global loads early (latency overlap) ----
        float4 p0, p1, p2, p3, q0, q1;
        if (hasNext) {
            const int kn = (kb + 1) * BK;
            const float4* sa = (const float4*)(aSrcBase + kn);
            p0 = sa[0]; p1 = sa[1]; p2 = sa[2]; p3 = sa[3];
            const float4* sw = (const float4*)(wSrcBase + (size_t)kn * N);
            q0 = sw[0]; q1 = sw[1];
            if (kb + 2 < NKSTEPS) {
                const int kp = (kb + 2) * BK;
                __builtin_prefetch(aSrcBase + kp, 0, 1);
                __builtin_prefetch(wSrcBase + (size_t)kp * N, 0, 1);
            }
        }

        // ---- fragment loads from current buffer (ds_load_b128 pairs) ----
        v16bf afr[2], bfr[2];
        #pragma unroll
        for (int f = 0; f < 2; ++f) {
            const int row = waveM * 32 + f * 16 + rA;
            const v8bf lo = *(const v8bf*)&As[cur][row * LDS_STRIDE + kh8];
            const v8bf hi = *(const v8bf*)&As[cur][row * LDS_STRIDE + 16 + kh8];
            #pragma unroll
            for (int i = 0; i < 8; ++i) { afr[f][i] = lo[i]; afr[f][i + 8] = hi[i]; }
        }
        #pragma unroll
        for (int g = 0; g < 2; ++g) {
            const int col = waveN * 32 + g * 16 + rA;
            const v8bf lo = *(const v8bf*)&Bs[cur][col * LDS_STRIDE + kh8];
            const v8bf hi = *(const v8bf*)&Bs[cur][col * LDS_STRIDE + 16 + kh8];
            #pragma unroll
            for (int i = 0; i < 8; ++i) { bfr[g][i] = lo[i]; bfr[g][i + 8] = hi[i]; }
        }

        // ---- 4 WMMAs ----
        #pragma unroll
        for (int f = 0; f < 2; ++f)
            #pragma unroll
            for (int g = 0; g < 2; ++g)
                acc[f][g] = __builtin_amdgcn_wmma_f32_16x16x32_bf16(
                    false, afr[f], false, bfr[g], (short)0, acc[f][g], false, false);

        // ---- store next tile into the other buffer ----
        if (hasNext) {
            const int nxt = cur ^ 1;
            uint4 u0 = { pk2bf(p0.x, p0.y), pk2bf(p0.z, p0.w),
                         pk2bf(p1.x, p1.y), pk2bf(p1.z, p1.w) };
            uint4 u1 = { pk2bf(p2.x, p2.y), pk2bf(p2.z, p2.w),
                         pk2bf(p3.x, p3.y), pk2bf(p3.z, p3.w) };
            *(uint4*)&As[nxt][aRow * LDS_STRIDE + aKg]     = u0;
            *(uint4*)&As[nxt][aRow * LDS_STRIDE + aKg + 8] = u1;
            Bs[nxt][(bNg + 0) * LDS_STRIDE + bK] = f32_to_bf16_bits(q0.x);
            Bs[nxt][(bNg + 1) * LDS_STRIDE + bK] = f32_to_bf16_bits(q0.y);
            Bs[nxt][(bNg + 2) * LDS_STRIDE + bK] = f32_to_bf16_bits(q0.z);
            Bs[nxt][(bNg + 3) * LDS_STRIDE + bK] = f32_to_bf16_bits(q0.w);
            Bs[nxt][(bNg + 4) * LDS_STRIDE + bK] = f32_to_bf16_bits(q1.x);
            Bs[nxt][(bNg + 5) * LDS_STRIDE + bK] = f32_to_bf16_bits(q1.y);
            Bs[nxt][(bNg + 6) * LDS_STRIDE + bK] = f32_to_bf16_bits(q1.z);
            Bs[nxt][(bNg + 7) * LDS_STRIDE + bK] = f32_to_bf16_bits(q1.w);
        }
        __syncthreads();
    }

    // ---- epilogue: C-layout VGPR r -> M = r (+8 for lanes 16..31), lane&15 -> N
    const int hi8 = (lane >= 16) ? 8 : 0;
    #pragma unroll
    for (int f = 0; f < 2; ++f) {
        #pragma unroll
        for (int g = 0; g < 2; ++g) {
            const int col = n0 + waveN * 32 + g * 16 + (lane & 15);
            const float bcol = bias[col];
            const int rowBase = m0 + waveM * 32 + f * 16 + hi8;
            float* cbase = &C[(size_t)rowBase * N + col];
            #pragma unroll
            for (int r = 0; r < 8; ++r)
                cbase[(size_t)r * N] = acc[f][g][r] + bcol;
        }
    }
}

// =====================================================================
// Zero-fill of the dense w output — non-temporal so the 512MB stream
// doesn't evict the (L2-resident) q/k/v working set.
// =====================================================================
__global__ void zero_f4(f4vec* __restrict__ p, size_t n4)
{
    size_t idx = (size_t)blockIdx.x * blockDim.x + threadIdx.x;
    const size_t stride = (size_t)gridDim.x * blockDim.x;
    const f4vec z = {0.f, 0.f, 0.f, 0.f};
    for (; idx < n4; idx += stride) __builtin_nontemporal_store(z, &p[idx]);
}

// =====================================================================
// Banded attention: one wave per (b,h,i) row; 8 waves per block.
// qv: [8192][2048] (cols 0..1023 = Q, 1024..2047 = V), kk: [8192][1024]
// =====================================================================
__global__ __launch_bounds__(256) void band_attn(
    const float* __restrict__ qv, const float* __restrict__ kk,
    const float* __restrict__ mfeats, const int* __restrict__ mask,
    float* __restrict__ outM, float* __restrict__ outW)
{
    const int lane = threadIdx.x & 31;
    const int row  = blockIdx.x * 8 + (threadIdx.x >> 5);   // ((b*H+h)*S + i)
    const int b = row >> 14;
    const int h = (row >> 10) & (NH - 1);
    const int i = row & (SEQ - 1);

    const int d0 = lane * 2;
    const float* Krow = kk + ((size_t)(b * SEQ + i) * ODIM + h * HD);
    const float k0 = Krow[d0], k1 = Krow[d0 + 1];

    float sc[NTAP];
    #pragma unroll
    for (int jj = 0; jj < NTAP; ++jj) {
        const int j = i - KHALF + jj;
        const bool ok = ((unsigned)j < (unsigned)SEQ) && (mask[b * SEQ + j] != 0);
        if (ok) {
            const float* Qrow = qv + ((size_t)(b * SEQ + j) * 2048 + h * HD);
            float p = k0 * Qrow[d0] + k1 * Qrow[d0 + 1];
            p = wave_reduce_sum(p);        // all lanes hold full dot
            sc[jj] = p * 0.125f;           // 1/sqrt(HD)
        } else {
            sc[jj] = -1.0e9f;
        }
    }

    float mx = sc[0];
    #pragma unroll
    for (int jj = 1; jj < NTAP; ++jj) mx = fmaxf(mx, sc[jj]);

    float* wrow = outW + (size_t)row * SEQ;

    if (mx > -1.0e8f) {
        float e[NTAP], den = 0.f;
        #pragma unroll
        for (int jj = 0; jj < NTAP; ++jj) {
            e[jj] = (sc[jj] < -1.0e8f) ? 0.f : __expf(sc[jj] - mx);
            den += e[jj];
        }
        const float inv = 1.0f / den;

        if (lane == 0) {
            #pragma unroll
            for (int jj = 0; jj < NTAP; ++jj) {
                const int j = i - KHALF + jj;
                if ((unsigned)j < (unsigned)SEQ) wrow[j] = e[jj] * inv;
            }
        }

        float a0 = 0.f, a1 = 0.f;
        #pragma unroll
        for (int jj = 0; jj < NTAP; ++jj) {
            const float wj = e[jj] * inv;
            if (wj > 0.f) {
                const int j = i - KHALF + jj;
                const float* Vrow = qv + ((size_t)(b * SEQ + j) * 2048 + 1024 + h * HD);
                a0 += wj * Vrow[d0];
                a1 += wj * Vrow[d0 + 1];
            }
        }
        const size_t mo = (size_t)(b * SEQ + i) * ODIM + h * HD + d0;
        outM[mo]     = mfeats[mo]     + a0;
        outM[mo + 1] = mfeats[mo + 1] + a1;
    } else {
        // degenerate all-masked row: reference softmax -> uniform 1/S over FULL row
        const float u = 1.0f / (float)SEQ;
        for (int c = lane; c < SEQ; c += 32) wrow[c] = u;
        float a0 = 0.f, a1 = 0.f;
        const float* Vbase = qv + ((size_t)(b * SEQ) * 2048 + 1024 + h * HD);
        for (int j = 0; j < SEQ; ++j) {
            const float* Vrow = Vbase + (size_t)j * 2048;
            a0 += Vrow[d0];
            a1 += Vrow[d0 + 1];
        }
        a0 *= u; a1 *= u;
        const size_t mo = (size_t)(b * SEQ + i) * ODIM + h * HD + d0;
        outM[mo]     = mfeats[mo]     + a0;
        outM[mo + 1] = mfeats[mo + 1] + a1;
    }
}

// =====================================================================
extern "C" void kernel_launch(void* const* d_in, const int* in_sizes, int n_in,
                              void* d_out, int out_size, void* d_ws, size_t ws_size,
                              hipStream_t stream)
{
    (void)in_sizes; (void)n_in; (void)out_size; (void)ws_size;
    const float* m_feats = (const float*)d_in[0];
    const int*   mask    = (const int*)d_in[1];
    const float* Wc      = (const float*)d_in[2];
    const float* bc      = (const float*)d_in[3];
    const float* Wv      = (const float*)d_in[4];
    const float* bv      = (const float*)d_in[5];

    float* outM = (float*)d_out;                                    // [8,1024,1024]
    float* outW = outM + (size_t)BATCH * SEQ * ODIM;                // [8,16,1024,1024]

    float* ws_qv = (float*)d_ws;                                    // [8192][2048]
    float* ws_k  = ws_qv + (size_t)(BATCH * SEQ) * (2 * ODIM);      // [8192][1024]

    // GEMM 1: QV = m_feats @ Wc + bc  (N = 2048)
    gemm_bf16_wmma<2 * ODIM><<<dim3((2 * ODIM) / BN, (BATCH * SEQ) / BM), 256, 0, stream>>>(
        m_feats, Wc, bc, ws_qv);
    // GEMM 2: K = m_feats @ Wv + bv   (N = 1024)
    gemm_bf16_wmma<ODIM><<<dim3(ODIM / BN, (BATCH * SEQ) / BM), 256, 0, stream>>>(
        m_feats, Wv, bv, ws_k);

    // zero the dense w output (out-of-band entries are exactly 0)
    const size_t n4 = (size_t)BATCH * NH * SEQ * SEQ / 4;
    zero_f4<<<32768, 256, 0, stream>>>((f4vec*)outW, n4);

    // banded softmax + weighted sum + residual add
    band_attn<<<(BATCH * NH * SEQ) / 8, 256, 0, stream>>>(
        ws_qv, ws_k, m_feats, mask, outM, outW);
}
